// RegularAttention_28862180229422
// MI455X (gfx1250) — matrix operand order repaired and checked by reading, not compiled, not run on hardware
//
#include <hip/hip_runtime.h>
#include <hip/hip_bf16.h>

typedef __attribute__((ext_vector_type(16))) __bf16 v16bf;
typedef __attribute__((ext_vector_type(8)))  __bf16 v8bf;
typedef __attribute__((ext_vector_type(8)))  float  v8f;
typedef __attribute__((ext_vector_type(4)))  unsigned int v4u;
typedef __attribute__((ext_vector_type(8)))  int v8i;
typedef __attribute__((ext_vector_type(4)))  int v4i;

#define DIMC   1024
#define SEQ    2048
#define BATCH  2
#define HEADS  16
#define DHEAD  64
#define SCALE  0.125f

union FragU { v16bf v; v8bf h[2]; };

__device__ __forceinline__ float half_reduce_max(float v) {
#pragma unroll
  for (int m = 8; m >= 1; m >>= 1) v = fmaxf(v, __shfl_xor(v, m, 32));
  return v;
}
__device__ __forceinline__ float half_reduce_sum(float v) {
#pragma unroll
  for (int m = 8; m >= 1; m >>= 1) v += __shfl_xor(v, m, 32);
  return v;
}

// ---------------------------------------------------------------------------
// Kernel 1: QKV = X[4096,1024] @ Wqkv[1024,3072], output bf16 scattered into
// Q/K/V buffers laid out [b, h, n, 64].
// Workgroup tile 128x128, K-step 32, 8 waves; wave tile 32x64 (2x4 WMMA).
// ---------------------------------------------------------------------------
__global__ __launch_bounds__(256) void qkv_gemm_kernel(
    const float* __restrict__ X, const float* __restrict__ W,
    __bf16* __restrict__ Qw, __bf16* __restrict__ Kw, __bf16* __restrict__ Vw) {
  __shared__ __bf16 As[128][40];  // [m][k] K-contiguous
  __shared__ __bf16 Bs[128][40];  // [n][k] transposed, K-contiguous

  const int t = threadIdx.x;
  const int lane = t & 31;
  const int wave = t >> 5;
  const int wm = wave >> 1;      // 0..3
  const int wn = wave & 1;       // 0..1
  const int m0 = blockIdx.y * 128;
  const int n0 = blockIdx.x * 128;
  const int hi = lane >> 4, lo = lane & 15;

  v8f zero = {};
  v8f acc[2][4];
#pragma unroll
  for (int i = 0; i < 2; i++)
#pragma unroll
    for (int j = 0; j < 4; j++) acc[i][j] = zero;

  for (int k0 = 0; k0 < DIMC; k0 += 32) {
    // stage A: 128x32 fp32 -> bf16
#pragma unroll
    for (int it = 0; it < 4; ++it) {
      int seg = t + it * 256;            // 1024 segs of 4 floats
      int row = seg >> 3;
      int col = (seg & 7) * 4;
      const float4 f = *(const float4*)&X[(size_t)(m0 + row) * DIMC + k0 + col];
      As[row][col + 0] = (__bf16)f.x;
      As[row][col + 1] = (__bf16)f.y;
      As[row][col + 2] = (__bf16)f.z;
      As[row][col + 3] = (__bf16)f.w;
    }
    // stage B transposed: 32x128 fp32 -> bf16 [n][k]
#pragma unroll
    for (int it = 0; it < 4; ++it) {
      int seg = t + it * 256;
      int k = seg >> 5;
      int nq = (seg & 31) * 4;
      const float4 f = *(const float4*)&W[(size_t)(k0 + k) * 3072 + n0 + nq];
      Bs[nq + 0][k] = (__bf16)f.x;
      Bs[nq + 1][k] = (__bf16)f.y;
      Bs[nq + 2][k] = (__bf16)f.z;
      Bs[nq + 3][k] = (__bf16)f.w;
    }
    __syncthreads();

    v16bf af[2], bfr[4];
#pragma unroll
    for (int i = 0; i < 2; i++) {
      int m = wm * 32 + i * 16 + lo;
      int ko = hi * 8;
      FragU u;
      u.h[0] = *(const v8bf*)&As[m][ko];
      u.h[1] = *(const v8bf*)&As[m][ko + 16];
      af[i] = u.v;
    }
#pragma unroll
    for (int j = 0; j < 4; j++) {
      int n = wn * 64 + j * 16 + lo;
      int ko = hi * 8;
      FragU u;
      u.h[0] = *(const v8bf*)&Bs[n][ko];
      u.h[1] = *(const v8bf*)&Bs[n][ko + 16];
      bfr[j] = u.v;
    }
#pragma unroll
    for (int i = 0; i < 2; i++)
#pragma unroll
      for (int j = 0; j < 4; j++)
        acc[i][j] = __builtin_amdgcn_wmma_f32_16x16x32_bf16(
            false, af[i], false, bfr[j], (short)0, acc[i][j], false, false);
    __syncthreads();
  }

  // epilogue: scatter bf16 into Q/K/V [b,h,n,64]
#pragma unroll
  for (int i = 0; i < 2; i++) {
#pragma unroll
    for (int j = 0; j < 4; j++) {
      int c = n0 + wn * 64 + j * 16 + lo;   // 0..3071
      int which = c >> 10;
      int cc = c & 1023;
      int h = cc >> 6, d = cc & 63;
      __bf16* dst = (which == 0) ? Qw : ((which == 1) ? Kw : Vw);
#pragma unroll
      for (int p = 0; p < 8; p++) {
        int m = m0 + wm * 32 + i * 16 + p + 8 * hi;  // 0..4095
        int b = m >> 11, nn = m & 2047;
        dst[(((size_t)(b * HEADS + h) * SEQ + nn) << 6) + d] = (__bf16)acc[i][j][p];
      }
    }
  }
}

// ---------------------------------------------------------------------------
// Kernel 2: flash attention per (b,h). 128 threads = 4 waves, each wave owns
// 16 query rows; KV tiles of 32. Q/K fragments straight from global bf16;
// V transposed through LDS; P reshaped C-layout -> A-layout through LDS.
// ---------------------------------------------------------------------------
__global__ __launch_bounds__(128) void attention_kernel(
    const __bf16* __restrict__ Qw, const __bf16* __restrict__ Kw,
    const __bf16* __restrict__ Vw, __bf16* __restrict__ Ow) {
  __shared__ __bf16 Pl[4][16][40];  // per-wave P slab
  __shared__ __bf16 Vt[64][40];     // V tile transposed [d][kvloc]

  const int t = threadIdx.x;
  const int lane = t & 31;
  const int wave = t >> 5;
  const int hi = lane >> 4, lo = lane & 15;
  const int bh = blockIdx.y;       // 0..31
  const int h = bh & (HEADS - 1);
  const int b = bh >> 4;
  const int q0 = blockIdx.x * 64 + wave * 16;
  const size_t base = (size_t)bh * SEQ * DHEAD;

  // Q fragments: two K(d)-chunks of 32
  v16bf qf[2];
#pragma unroll
  for (int c = 0; c < 2; c++) {
    int row = q0 + lo;
    int dof = c * 32 + hi * 8;
    FragU u;
    u.h[0] = *(const v8bf*)&Qw[base + (size_t)row * DHEAD + dof];
    u.h[1] = *(const v8bf*)&Qw[base + (size_t)row * DHEAD + dof + 16];
    qf[c] = u.v;
  }

  v8f zero = {};
  v8f o[4] = {zero, zero, zero, zero};
  float m8[8], l8[8];
#pragma unroll
  for (int p = 0; p < 8; p++) { m8[p] = -3.0e38f; l8[p] = 0.0f; }

  for (int kv = 0; kv < SEQ; kv += 32) {
    // prefetch next K tile rows (speculative, no counter cost)
    if (kv + 32 < SEQ)
      __builtin_prefetch(&Kw[base + (size_t)(kv + 32 + lane) * DHEAD], 0, 1);

    // cooperative stage of V tile 32x64 -> transposed LDS [d][kvloc]
#pragma unroll
    for (int it = 0; it < 2; ++it) {
      int seg = t + it * 128;          // 256 segs of 8 bf16
      int row = seg >> 3;              // kv row 0..31
      int ds0 = (seg & 7) * 8;         // d start
      v8bf v = *(const v8bf*)&Vw[base + (size_t)(kv + row) * DHEAD + ds0];
#pragma unroll
      for (int e = 0; e < 8; e++) Vt[ds0 + e][row] = v[e];
    }
    __syncthreads();

    // S = Q @ K^T for two 16-key sub-tiles (contract over d=64 in 2 WMMAs)
    v8f s[2];
#pragma unroll
    for (int jj = 0; jj < 2; jj++) {
      v8f sacc = zero;
#pragma unroll
      for (int c = 0; c < 2; c++) {
        int key = kv + jj * 16 + lo;
        int dof = c * 32 + hi * 8;
        FragU u;
        u.h[0] = *(const v8bf*)&Kw[base + (size_t)key * DHEAD + dof];
        u.h[1] = *(const v8bf*)&Kw[base + (size_t)key * DHEAD + dof + 16];
        sacc = __builtin_amdgcn_wmma_f32_16x16x32_bf16(
            false, qf[c], false, u.v, (short)0, sacc, false, false);
      }
      s[jj] = sacc;
    }

    // online softmax per accumulator row
    float alpha[8];
#pragma unroll
    for (int p = 0; p < 8; p++) {
      float s0 = s[0][p] * SCALE;
      float s1 = s[1][p] * SCALE;
      float cmax = half_reduce_max(fmaxf(s0, s1));
      float mnew = fmaxf(m8[p], cmax);
      float a = __expf(m8[p] - mnew);
      float p0 = __expf(s0 - mnew);
      float p1 = __expf(s1 - mnew);
      l8[p] = l8[p] * a + half_reduce_sum(p0 + p1);
      m8[p] = mnew;
      alpha[p] = a;
      int row = p + 8 * hi;
      Pl[wave][row][lo] = (__bf16)p0;
      Pl[wave][row][16 + lo] = (__bf16)p1;
    }
#pragma unroll
    for (int nt = 0; nt < 4; nt++)
#pragma unroll
      for (int p = 0; p < 8; p++) o[nt][p] *= alpha[p];

    // P C-layout -> A-fragment via per-wave LDS slab (same-wave RAW, in-order DS)
    FragU up;
    up.h[0] = *(const v8bf*)&Pl[wave][lo][hi * 8];
    up.h[1] = *(const v8bf*)&Pl[wave][lo][hi * 8 + 16];
    v16bf pf = up.v;

    // O += P @ V  (4 d-tiles of 16)
#pragma unroll
    for (int nt = 0; nt < 4; nt++) {
      int dcol = nt * 16 + lo;
      int ko = hi * 8;
      FragU uv;
      uv.h[0] = *(const v8bf*)&Vt[dcol][ko];
      uv.h[1] = *(const v8bf*)&Vt[dcol][ko + 16];
      o[nt] = __builtin_amdgcn_wmma_f32_16x16x32_bf16(
          false, pf, false, uv.v, (short)0, o[nt], false, false);
    }
    __syncthreads();
  }

  // epilogue: O /= l, write bf16 to Ow [b][n][h*64+d]
#pragma unroll
  for (int p = 0; p < 8; p++) {
    float inv = 1.0f / l8[p];
    int q = q0 + p + 8 * hi;
#pragma unroll
    for (int nt = 0; nt < 4; nt++) {
      int d = nt * 16 + lo;
      Ow[((size_t)(b * SEQ + q) << 10) + h * DHEAD + d] = (__bf16)(o[nt][p] * inv);
    }
  }
}

// ---------------------------------------------------------------------------
// Kernel 3: out = Ow[4096,1024] @ Wout[1024,1024] + bias, fp32 output.
// A operand (already bf16) is staged via the Tensor Data Mover:
//   - 2D tile 32 (dim0) x 128 (dim1) elements, row = 64B, tensor stride 1024.
//   - TDM LDS padding (pad_interval=16 DW, pad_amount=4 DW) reproduces the
//     [128][40] padded LDS layout the fragment loaders expect.
//   - Issued from wave 0 only (tensor ops ignore EXEC), completion via
//     s_wait_tensorcnt 0 then workgroup barrier.
// ---------------------------------------------------------------------------
__global__ __launch_bounds__(256) void out_gemm_kernel(
    const __bf16* __restrict__ A, const float* __restrict__ W,
    const float* __restrict__ bias, float* __restrict__ out) {
  __shared__ __bf16 As[128][40];
  __shared__ __bf16 Bs[128][40];

  const int t = threadIdx.x;
  const int lane = t & 31;
  const int wave = t >> 5;
  const int wm = wave >> 1;
  const int wn = wave & 1;
  const int m0 = blockIdx.y * 128;
  const int n0 = blockIdx.x * 128;
  const int hi = lane >> 4, lo = lane & 15;

  // LDS byte offset of As (low 32 bits of the generic shared address)
  const unsigned int lds_as = (unsigned int)(size_t)&As[0][0];

  v8f zero = {};
  v8f acc[2][4];
#pragma unroll
  for (int i = 0; i < 2; i++)
#pragma unroll
    for (int j = 0; j < 4; j++) acc[i][j] = zero;

  for (int k0 = 0; k0 < DIMC; k0 += 32) {
    // --- TDM: async DMA of the 128x32 bf16 A tile into padded LDS ---
    if (wave == 0) {
      unsigned long long ga =
          (unsigned long long)(size_t)&A[(size_t)m0 * DIMC + k0];
      v4u g0;
      g0[0] = 1u;                                   // count=1, user mode
      g0[1] = lds_as;                               // lds_addr
      g0[2] = (unsigned int)ga;                     // global_addr[31:0]
      g0[3] = (unsigned int)((ga >> 32) & 0x1FFFFFFull) | (2u << 30); // type=2
      v8i g1;
      g1[0] = (1 << 16)        /* data_size = 2 bytes */
            | (1 << 20)        /* pad_enable */
            | (3 << 22)        /* pad_interval: 16 DWORDs (64B row) */
            | (3 << 25);       /* pad_amount: 4 DWORDs (16B pad) */
      g1[1] = (DIMC & 0xFFFF) << 16;                       // tensor_dim0 lo16
      g1[2] = ((DIMC >> 16) & 0xFFFF) | ((4096 & 0xFFFF) << 16); // d0 hi|d1 lo
      g1[3] = ((4096 >> 16) & 0xFFFF) | (32 << 16);        // d1 hi|tile_dim0=32
      g1[4] = 128;                                         // tile_dim1=128
      g1[5] = DIMC;                                        // dim0_stride lo32
      g1[6] = 0;                                           // stride hi bits
      g1[7] = 0;
      v4i g2 = {0, 0, 0, 0};
      v4i g3 = {0, 0, 0, 0};
      v8i g4 = {0, 0, 0, 0, 0, 0, 0, 0};  // extra operand group (clang-23 form)
      __builtin_amdgcn_tensor_load_to_lds(g0, g1, g2, g3, g4, 0);
      __builtin_amdgcn_s_wait_tensorcnt(0);
    }
    // stage B transposed fp32 -> bf16 (conversion => must go through VALU)
#pragma unroll
    for (int it = 0; it < 4; ++it) {
      int seg = t + it * 256;
      int k = seg >> 5;
      int nq = (seg & 31) * 4;
      const float4 f = *(const float4*)&W[(size_t)(k0 + k) * DIMC + n0 + nq];
      Bs[nq + 0][k] = (__bf16)f.x;
      Bs[nq + 1][k] = (__bf16)f.y;
      Bs[nq + 2][k] = (__bf16)f.z;
      Bs[nq + 3][k] = (__bf16)f.w;
    }
    __syncthreads();

    v16bf af[2], bfr[4];
#pragma unroll
    for (int i = 0; i < 2; i++) {
      int m = wm * 32 + i * 16 + lo;
      int ko = hi * 8;
      FragU u;
      u.h[0] = *(const v8bf*)&As[m][ko];
      u.h[1] = *(const v8bf*)&As[m][ko + 16];
      af[i] = u.v;
    }
#pragma unroll
    for (int j = 0; j < 4; j++) {
      int n = wn * 64 + j * 16 + lo;
      int ko = hi * 8;
      FragU u;
      u.h[0] = *(const v8bf*)&Bs[n][ko];
      u.h[1] = *(const v8bf*)&Bs[n][ko + 16];
      bfr[j] = u.v;
    }
#pragma unroll
    for (int i = 0; i < 2; i++)
#pragma unroll
      for (int j = 0; j < 4; j++)
        acc[i][j] = __builtin_amdgcn_wmma_f32_16x16x32_bf16(
            false, af[i], false, bfr[j], (short)0, acc[i][j], false, false);
    __syncthreads();
  }

#pragma unroll
  for (int j = 0; j < 4; j++) {
    int c = n0 + wn * 64 + j * 16 + lo;
    float bi = bias[c];
#pragma unroll
    for (int i = 0; i < 2; i++) {
#pragma unroll
      for (int p = 0; p < 8; p++) {
        int m = m0 + wm * 32 + i * 16 + p + 8 * hi;
        out[(size_t)m * DIMC + c] = acc[i][j][p] + bi;
      }
    }
  }
}

// ---------------------------------------------------------------------------
extern "C" void kernel_launch(void* const* d_in, const int* in_sizes, int n_in,
                              void* d_out, int out_size, void* d_ws, size_t ws_size,
                              hipStream_t stream) {
  const float* X    = (const float*)d_in[0];  // [2,2048,1024]
  const float* Wqkv = (const float*)d_in[1];  // [1024,3072]
  const float* Wout = (const float*)d_in[2];  // [1024,1024]
  const float* bout = (const float*)d_in[3];  // [1024]
  float* out = (float*)d_out;                 // [2,2048,1024]

  const size_t QE = (size_t)BATCH * HEADS * SEQ * DHEAD;  // 4M elems
  __bf16* Qw = (__bf16*)d_ws;
  __bf16* Kw = Qw + QE;
  __bf16* Vw = Kw + QE;
  __bf16* Ow = Vw + QE;  // [b][n][1024] bf16

  qkv_gemm_kernel<<<dim3(3072 / 128, 4096 / 128), 256, 0, stream>>>(X, Wqkv, Qw, Kw, Vw);
  attention_kernel<<<dim3(SEQ / 64, BATCH * HEADS), 128, 0, stream>>>(Qw, Kw, Vw, Ow);
  out_gemm_kernel<<<dim3(DIMC / 128, 4096 / 128), 256, 0, stream>>>(Ow, Wout, bout, out);
}
